// Encoder_84748294685286
// MI455X (gfx1250) — matrix operand order repaired and checked by reading, not compiled, not run on hardware
//
#include <hip/hip_runtime.h>
#include <hip/hip_bf16.h>

typedef __bf16 bf16;
typedef __attribute__((ext_vector_type(16))) __bf16 v16bf;
typedef __attribute__((ext_vector_type(8)))  __bf16 v8bf;
typedef __attribute__((ext_vector_type(8)))  float  v8f;

constexpr int BATCH = 128;
constexpr int TSTEPS = 512;
constexpr int FDIM = 64;
constexpr int U1DIM = 256;
constexpr int U2DIM = 128;
constexpr int NWG_L1 = 16;           // unit tiles for layer 1 (256/16)
constexpr int NWG_L2 = 8;            // unit tiles for layer 2 (128/16)
constexpr int NWG = NWG_L1 + NWG_L2; // one pipelined persistent grid

// ---------------------------------------------------------------------------
// Monotonic grid-wide barrier (persistent small grid; arrive + spin).
// bar[0] = arrival counter, bar[1] = released phase. Zeroed each launch.
// ---------------------------------------------------------------------------
__device__ __forceinline__ void grid_sync(unsigned* bar, unsigned nwg, unsigned step) {
  __threadfence();
  __syncthreads();
  if (threadIdx.x == 0) {
    unsigned prev = __hip_atomic_fetch_add(&bar[0], 1u, __ATOMIC_ACQ_REL,
                                           __HIP_MEMORY_SCOPE_AGENT);
    if (prev == (step + 1u) * nwg - 1u) {
      __hip_atomic_store(&bar[1], step + 1u, __ATOMIC_RELEASE,
                         __HIP_MEMORY_SCOPE_AGENT);
    } else {
      while (__hip_atomic_load(&bar[1], __ATOMIC_ACQUIRE,
                               __HIP_MEMORY_SCOPE_AGENT) <= step) {
        __builtin_amdgcn_s_sleep(2);
      }
    }
  }
  __syncthreads();
  __threadfence();
}

// ---------------------------------------------------------------------------
// Pack fused weights [Wx; Uh] transposed to bf16 [4*UNITS][KX+UNITS] (n-major)
// so each lane's WMMA B-fragment is a contiguous 32B read.
// ---------------------------------------------------------------------------
__global__ void k_pack_weights(const float* __restrict__ W1g, const float* __restrict__ U1g,
                               const float* __restrict__ W2g, const float* __restrict__ U2g,
                               bf16* __restrict__ Wc1, bf16* __restrict__ Wc2) {
  const int total1 = 4 * U1DIM * (FDIM + U1DIM);   // 1024*320
  const int total2 = 4 * U2DIM * (U1DIM + U2DIM);  // 512*384
  for (int i = blockIdx.x * blockDim.x + threadIdx.x; i < total1 + total2;
       i += gridDim.x * blockDim.x) {
    if (i < total1) {
      int n = i / 320, k = i - n * 320;
      float v = (k < FDIM) ? W1g[(size_t)k * 1024 + n]
                           : U1g[(size_t)(k - FDIM) * 1024 + n];
      Wc1[(size_t)n * 320 + k] = (bf16)v;
    } else {
      int j = i - total1;
      int n = j / 384, k = j - n * 384;
      float v = (k < U1DIM) ? W2g[(size_t)k * 512 + n]
                            : U2g[(size_t)(k - U1DIM) * 512 + n];
      Wc2[(size_t)n * 384 + k] = (bf16)v;
    }
  }
}

// x [B][T][F] fp32  ->  xbf [T][B][F] bf16 (time-major for the recurrence)
__global__ void k_pack_x(const float* __restrict__ x, bf16* __restrict__ xbf) {
  const int total = BATCH * TSTEPS * FDIM;
  for (int i = blockIdx.x * blockDim.x + threadIdx.x; i < total;
       i += gridDim.x * blockDim.x) {
    int f = i % FDIM;
    int t = (i / FDIM) % TSTEPS;
    int b = i / (FDIM * TSTEPS);
    xbf[((size_t)t * BATCH + b) * FDIM + f] =
        (bf16)x[((size_t)b * TSTEPS + t) * FDIM + f];
  }
}

// zero h1_seq[0], h2_seq[0], and the barrier block (ws is poisoned 0xAA)
__global__ void k_init(bf16* __restrict__ h1_0, bf16* __restrict__ h2_0,
                       unsigned* __restrict__ bar) {
  int i = blockIdx.x * blockDim.x + threadIdx.x;
  if (i < BATCH * U1DIM) h1_0[i] = (bf16)0.0f;
  if (i < BATCH * U2DIM) h2_0[i] = (bf16)0.0f;
  if (i < 8) bar[i] = 0u;
}

// ---------------------------------------------------------------------------
// One pipelined LSTM layer "lane" inside the fused persistent kernel.
// One wave = one 16x16 (batch, unit) tile. All 8 waves of a WG share one
// 64-column weight slab in LDS (4 gates x 16 unit cols x KTOT).
// SKEW = phase offset of this layer in the software pipeline.
// ---------------------------------------------------------------------------
template <int KX, int UNITS, int SKEW>
__device__ __forceinline__ void run_layer(
    const bf16* __restrict__ xin,    // [T][BATCH][KX]
    const bf16* __restrict__ Wc,     // [4*UNITS][KX+UNITS]
    const float* __restrict__ bias,  // [4*UNITS] gate order i,f,g,o
    bf16* __restrict__ hseq,         // [T+1][BATCH][UNITS], slot 0 zeroed
    float* __restrict__ outf,        // final-h fp32 [BATCH][UNITS] or nullptr
    unsigned* __restrict__ bar, int T, bf16* lds_w, int un) {
  constexpr int KTOT = KX + UNITS;
  constexpr int NCH = KTOT / 32;  // k-chunks of 32 per WMMA
  constexpr int CHX = KX / 32;    // chunks sourced from x vs h

  const int wv = threadIdx.x >> 5;  // batch tile index (8 waves)
  const int lane = threadIdx.x & 31;
  const int lrow = lane & 15;       // A-row within tile / B-col / C-N
  const int hiH = lane >> 4;
  const int aoff = hiH ? 16 : 0;    // A fragment K base (contiguous, matches B)
  const int koff = hiH ? 16 : 0;    // B fragment K base (contiguous, matches A)
  const int row = wv * 16 + lrow;   // global batch row this lane loads

  // Stage this WG's weight slab into LDS once; reused for all T steps.
  for (int i = threadIdx.x; i < 64 * KTOT; i += 256) {
    int col = i / KTOT, k = i - col * KTOT;
    int g = col >> 4, j = col & 15;
    lds_w[i] = Wc[(size_t)(g * UNITS + un * 16 + j) * KTOT + k];
  }
  __syncthreads();

  float bv[4];
#pragma unroll
  for (int g = 0; g < 4; ++g) bv[g] = bias[g * UNITS + un * 16 + lrow];

  v8f creg = {};  // fp32 cell state, register-resident across all steps

  for (int s = 0; s <= T; ++s) {
    const int t = s - SKEW;
    if (t >= 0 && t < T) {
      const bf16* xr = xin + ((size_t)t * BATCH + row) * KX;
      const bf16* hr = hseq + ((size_t)t * BATCH + row) * UNITS;
      if (t + 1 < T)
        __builtin_prefetch(xin + ((size_t)(t + 1) * BATCH + row) * KX, 0, 1);

      v8f acc[4] = {v8f{}, v8f{}, v8f{}, v8f{}};

#pragma unroll
      for (int c = 0; c < NCH; ++c) {
        const bf16* src = (c < CHX) ? (xr + c * 32) : (hr + (c - CHX) * 32);
        v8bf lo = *(const v8bf*)(src + aoff);
        v8bf hi = *(const v8bf*)(src + 8 + aoff);
        v16bf af;
#pragma unroll
        for (int e = 0; e < 8; ++e) { af[e] = lo[e]; af[8 + e] = hi[e]; }
        // Load all four gate fragments up front so the scheduler can issue
        // all 8 ds_loads and pay a single dscnt wait before the WMMA quartet.
        const bf16* wbase = lds_w + lrow * KTOT + c * 32 + koff;
        v16bf bfi = *(const v16bf*)(wbase + 0 * 16 * KTOT);
        v16bf bff = *(const v16bf*)(wbase + 1 * 16 * KTOT);
        v16bf bfg = *(const v16bf*)(wbase + 2 * 16 * KTOT);
        v16bf bfo = *(const v16bf*)(wbase + 3 * 16 * KTOT);
        acc[0] = __builtin_amdgcn_wmma_f32_16x16x32_bf16(false, af, false, bfi,
                                                         (short)0, acc[0], false, false);
        acc[1] = __builtin_amdgcn_wmma_f32_16x16x32_bf16(false, af, false, bff,
                                                         (short)0, acc[1], false, false);
        acc[2] = __builtin_amdgcn_wmma_f32_16x16x32_bf16(false, af, false, bfg,
                                                         (short)0, acc[2], false, false);
        acc[3] = __builtin_amdgcn_wmma_f32_16x16x32_bf16(false, af, false, bfo,
                                                         (short)0, acc[3], false, false);
      }

      // Gates: i,f,o sigmoid; g relu; c' = f*c + i*g; h = o*relu(c')
      bf16* hw = hseq + ((size_t)(t + 1) * BATCH + wv * 16) * UNITS + un * 16 + lrow;
#pragma unroll
      for (int r = 0; r < 8; ++r) {
        const int M = r + (hiH ? 8 : 0);
        float zi = acc[0][r] + bv[0];
        float zf = acc[1][r] + bv[1];
        float zg = acc[2][r] + bv[2];
        float zo = acc[3][r] + bv[3];
        float ig = 1.0f / (1.0f + __expf(-zi));
        float fg = 1.0f / (1.0f + __expf(-zf));
        float gg = fmaxf(zg, 0.0f);
        float og = 1.0f / (1.0f + __expf(-zo));
        float cn = fg * creg[r] + ig * gg;
        creg[r] = cn;
        float hn = og * fmaxf(cn, 0.0f);
        hw[(size_t)M * UNITS] = (bf16)hn;
        if (outf != nullptr && t == T - 1)
          outf[(size_t)(wv * 16 + M) * UNITS + un * 16 + lrow] = hn;
      }
    }
    if (s < T) grid_sync(bar, NWG, (unsigned)s);
  }
}

// ---------------------------------------------------------------------------
// Fused persistent kernel: 24 WGs. WGs 0..15 run layer 1 at phase s (t = s);
// WGs 16..23 run layer 2 one step behind (t = s-1). One shared barrier.
// Layer-2's read of h1[t+1] at phase s was produced by layer-1 at phase s-1
// and made visible by barrier s-1.
// ---------------------------------------------------------------------------
__global__ __launch_bounds__(256) void lstm_fused(
    const bf16* __restrict__ xbf, const bf16* __restrict__ Wc1,
    const float* __restrict__ b1, bf16* __restrict__ h1,
    const bf16* __restrict__ Wc2, const float* __restrict__ b2,
    bf16* __restrict__ h2, float* __restrict__ out,
    unsigned* __restrict__ bar, int T) {
  __shared__ bf16 lds_w[64 * (U1DIM + U2DIM)];  // 48KB, max of both layers
  if (blockIdx.x < NWG_L1) {
    run_layer<FDIM, U1DIM, 0>(xbf, Wc1, b1, h1, nullptr, bar, T, lds_w,
                              blockIdx.x);
  } else {
    run_layer<U1DIM, U2DIM, 1>(h1 + (size_t)BATCH * U1DIM, Wc2, b2, h2, out,
                               bar, T, lds_w, blockIdx.x - NWG_L1);
  }
}

// ---------------------------------------------------------------------------
extern "C" void kernel_launch(void* const* d_in, const int* in_sizes, int n_in,
                              void* d_out, int out_size, void* d_ws, size_t ws_size,
                              hipStream_t stream) {
  const float* x  = (const float*)d_in[0];
  const float* W1 = (const float*)d_in[1];
  const float* U1 = (const float*)d_in[2];
  const float* b1 = (const float*)d_in[3];
  const float* W2 = (const float*)d_in[4];
  const float* U2 = (const float*)d_in[5];
  const float* b2 = (const float*)d_in[6];
  float* out = (float*)d_out;

  char* p = (char*)d_ws;
  auto carve = [&](size_t bytes) {
    char* r = p;
    p += (bytes + 255) & ~(size_t)255;
    return r;
  };
  unsigned* bar = (unsigned*)carve(256);
  bf16* Wc1 = (bf16*)carve((size_t)4 * U1DIM * (FDIM + U1DIM) * sizeof(bf16));
  bf16* Wc2 = (bf16*)carve((size_t)4 * U2DIM * (U1DIM + U2DIM) * sizeof(bf16));
  bf16* xbf = (bf16*)carve((size_t)TSTEPS * BATCH * FDIM * sizeof(bf16));
  bf16* h1  = (bf16*)carve((size_t)(TSTEPS + 1) * BATCH * U1DIM * sizeof(bf16));
  bf16* h2  = (bf16*)carve((size_t)(TSTEPS + 1) * BATCH * U2DIM * sizeof(bf16));
  (void)ws_size; (void)in_sizes; (void)n_in; (void)out_size;

  k_pack_weights<<<512, 256, 0, stream>>>(W1, U1, W2, U2, Wc1, Wc2);
  k_pack_x<<<1024, 256, 0, stream>>>(x, xbf);
  k_init<<<128, 256, 0, stream>>>(h1, h2, bar);

  // Single pipelined persistent kernel: 16 L1 WGs + 8 L2 WGs, one-step skew.
  lstm_fused<<<NWG, 256, 0, stream>>>(xbf, Wc1, b1, h1, Wc2, b2, h2, out, bar,
                                      TSTEPS);
}
